// GraniteMoeHybridMoE_43963285242733
// MI455X (gfx1250) — matrix operand order repaired and checked by reading, not compiled, not run on hardware
//
#include <hip/hip_runtime.h>
#include <hip/hip_bf16.h>

// Problem constants (from reference)
#define B_    2
#define L_    1024
#define D_    1536
#define H_    512
#define E_    64
#define KTOP  6
#define NTOK  (B_ * L_)
#define CAP   2048          // worst-case tokens per expert
#define TT    32            // tokens per expert tile (2 x 16-row WMMA subtiles)

typedef __attribute__((ext_vector_type(16))) __bf16         v16bf;
typedef __attribute__((ext_vector_type(8)))  float          v8f;
typedef __attribute__((ext_vector_type(8)))  unsigned int   v8u;
typedef __attribute__((ext_vector_type(4)))  unsigned int   v4u;

// ---- fp32 -> bf16 pack: single v_perm_b32 per pair (truncation) -------------
__device__ __forceinline__ unsigned int pkbf(float f0, float f1) {
    // result = { f1[31:16], f0[31:16] }
    return __builtin_amdgcn_perm(__float_as_uint(f1), __float_as_uint(f0), 0x07060302u);
}
__device__ __forceinline__ unsigned short f2bf_rn(float f) {   // epilogue only
    return (unsigned short)((__float_as_uint(f) + 0x8000u) >> 16);
}
__device__ __forceinline__ v16bf cvt16(float4 a, float4 b, float4 c, float4 d) {
    v8u t;
    t[0] = pkbf(a.x, a.y); t[1] = pkbf(a.z, a.w);
    t[2] = pkbf(b.x, b.y); t[3] = pkbf(b.z, b.w);
    t[4] = pkbf(c.x, c.y); t[5] = pkbf(c.z, c.w);
    t[6] = pkbf(d.x, d.y); t[7] = pkbf(d.z, d.w);
    return __builtin_bit_cast(v16bf, t);
}
// A fragment straight from LDS (bf16 already packed): two ds_load_b128
__device__ __forceinline__ v16bf ldsA(const unsigned short* ar, int k, int hi) {
    v4u c0 = *(const v4u*)(ar + k + hi * 8);
    v4u c1 = *(const v4u*)(ar + k + hi * 8 + 16);
    v8u t;
    t[0] = c0[0]; t[1] = c0[1]; t[2] = c0[2]; t[3] = c0[3];
    t[4] = c1[0]; t[5] = c1[1]; t[6] = c1[2]; t[7] = c1[3];
    return __builtin_bit_cast(v16bf, t);
}
#define WMMA_BF16(A, B, C) __builtin_amdgcn_wmma_f32_16x16x32_bf16( \
    false, (A), false, (B), (short)0, (C), false, false)

// ============================== zero-init =====================================
__global__ void moe_zero(float* __restrict__ out, int* __restrict__ counts, int n) {
    int i = blockIdx.x * blockDim.x + threadIdx.x;
    if (i < n) out[i] = 0.0f;
    if (i < E_) counts[i] = 0;
}

// ============================== router ========================================
// One wave per token: 32 lanes x 2 experts = 64 logits; lane 0 does top-6 +
// softmax and scatters the token into per-expert lists.
__global__ __launch_bounds__(256) void moe_router(
    const float* __restrict__ x, const float* __restrict__ w_router,
    int* __restrict__ counts, int* __restrict__ tok_ids, float* __restrict__ gatew)
{
    __shared__ float s_lg[8][E_];
    const int wv = threadIdx.x >> 5;
    const int lane = threadIdx.x & 31;
    const int token = blockIdx.x * 8 + wv;
    const float* xr = x + (size_t)token * D_;

    #pragma unroll
    for (int half = 0; half < 2; ++half) {
        const int e = lane + half * 32;
        const float* wr = w_router + (size_t)e * D_;
        float acc = 0.0f;
        for (int d = 0; d < D_; d += 4) {
            float4 xv = *(const float4*)(xr + d);
            float4 wf = *(const float4*)(wr + d);
            acc += xv.x * wf.x + xv.y * wf.y + xv.z * wf.z + xv.w * wf.w;
        }
        s_lg[wv][e] = acc;
    }
    __syncthreads();

    if (lane == 0) {
        float bv[KTOP]; int bi[KTOP];
        unsigned long long chosen = 0ull;
        #pragma unroll
        for (int s = 0; s < KTOP; ++s) {
            float best = -__builtin_inff(); int besti = 0;
            for (int e = 0; e < E_; ++e) {
                float v = s_lg[wv][e];
                if (!((chosen >> e) & 1ull) && v > best) { best = v; besti = e; }
            }
            chosen |= (1ull << besti);
            bv[s] = best; bi[s] = besti;
        }
        float mx = bv[0];           // first pick is the max
        float ex[KTOP]; float sum = 0.0f;
        #pragma unroll
        for (int s = 0; s < KTOP; ++s) { ex[s] = __expf(bv[s] - mx); sum += ex[s]; }
        const float inv = 1.0f / sum;
        #pragma unroll
        for (int s = 0; s < KTOP; ++s) {
            const int e = bi[s];
            const int pos = atomicAdd(&counts[e], 1);
            tok_ids[(size_t)e * CAP + pos] = token;
            gatew[(size_t)e * CAP + pos]   = ex[s] * inv;
        }
    }
}

// ============================== expert MLP ====================================
// blockIdx.y = expert, blockIdx.x = 32-token tile of that expert's token list.
// 8 waves, each owning 4 N-subtiles (GEMM1) / 12 N-subtiles (GEMM2) across two
// 16-row M-subtiles. Weights streamed fp32 -> v_perm truncate-pack -> bf16 WMMA
// with f32 accumulation; each B fragment feeds two WMMAs.
__global__ __launch_bounds__(256) void moe_expert(
    const float* __restrict__ x,
    const float* __restrict__ w_gate, const float* __restrict__ w_up,
    const float* __restrict__ w_down,
    const int* __restrict__ counts, const int* __restrict__ tok_ids,
    const float* __restrict__ gatew, float* __restrict__ out)
{
    const int e    = blockIdx.y;
    const int tile = blockIdx.x;
    const int ne   = counts[e];
    if (tile * TT >= ne) return;

    __shared__ int   s_tok[TT];
    __shared__ float s_gate[TT];
    __shared__ __align__(16) unsigned short s_a[TT][H_ + 8];  // bf16 acts, padded

    const int tid  = threadIdx.x;
    const int w    = tid >> 5;
    const int lane = tid & 31;
    const int lo   = lane & 15;     // M for A, N for B/C
    const int hi   = lane >> 4;     // K-base selector

    if (tid < TT) {
        const int r = tile * TT + tid;
        const int valid = (r < ne);
        s_tok[tid]  = tok_ids[(size_t)e * CAP + (valid ? r : 0)];
        s_gate[tid] = valid ? gatew[(size_t)e * CAP + r] : 0.0f;  // gate=0 -> pad rows contribute 0
    }
    __syncthreads();

    const float* wg = w_gate + (size_t)e * H_ * D_;
    const float* wu = w_up   + (size_t)e * H_ * D_;

    // ---------------- GEMM1: gate & up projections over K = D ----------------
    v8f accG0[4] = {}; v8f accG1[4] = {};
    v8f accU0[4] = {}; v8f accU1[4] = {};
    const float* xr0 = x + (size_t)s_tok[lo]      * D_ + hi * 8;
    const float* xr1 = x + (size_t)s_tok[16 + lo] * D_ + hi * 8;

    for (int k = 0; k < D_; k += 32) {
        if (k + 32 < D_) {
            __builtin_prefetch(xr0 + k + 32, 0, 3);
            __builtin_prefetch(xr1 + k + 32, 0, 3);
        }
        // Two A fragments (rows 0-15 and 16-31 of the token tile)
        const v16bf af0 = cvt16(*(const float4*)(xr0 + k),      *(const float4*)(xr0 + k + 4),
                                *(const float4*)(xr0 + k + 16), *(const float4*)(xr0 + k + 20));
        const v16bf af1 = cvt16(*(const float4*)(xr1 + k),      *(const float4*)(xr1 + k + 4),
                                *(const float4*)(xr1 + k + 16), *(const float4*)(xr1 + k + 20));

        #pragma unroll
        for (int j = 0; j < 4; ++j) {
            const int h0 = (w * 4 + j) * 16;
            const size_t roff = (size_t)(h0 + lo) * D_ + k + hi * 8;
            const float4* bgp = (const float4*)(wg + roff);
            const v16bf bg = cvt16(bgp[0], bgp[1], bgp[4], bgp[5]);
            accG0[j] = WMMA_BF16(af0, bg, accG0[j]);
            accG1[j] = WMMA_BF16(af1, bg, accG1[j]);
            const float4* bup = (const float4*)(wu + roff);
            const v16bf bu = cvt16(bup[0], bup[1], bup[4], bup[5]);
            accU0[j] = WMMA_BF16(af0, bu, accU0[j]);
            accU1[j] = WMMA_BF16(af1, bu, accU1[j]);
        }
    }

    // ------------- epilogue: a = silu(g) * u * gate -> LDS (bf16) -------------
    #pragma unroll
    for (int j = 0; j < 4; ++j) {
        const int h0 = (w * 4 + j) * 16;
        #pragma unroll
        for (int r = 0; r < 8; ++r) {
            const int M0 = r + hi * 8;                // C/D layout: row = vgpr + 8*hi
            float g = accG0[j][r], u = accU0[j][r];
            s_a[M0][h0 + lo] = f2bf_rn((g / (1.0f + __expf(-g))) * u * s_gate[M0]);
            const int M1 = 16 + M0;
            g = accG1[j][r]; u = accU1[j][r];
            s_a[M1][h0 + lo] = f2bf_rn((g / (1.0f + __expf(-g))) * u * s_gate[M1]);
        }
    }
    __syncthreads();

    // ---------------- GEMM2: down projection over K = H ----------------------
    const float* wd = w_down + (size_t)e * D_ * H_;
    const unsigned short* ar0 = &s_a[lo][0];
    const unsigned short* ar1 = &s_a[16 + lo][0];

    for (int j = 0; j < 12; ++j) {                    // 96 N-tiles / 8 waves
        const int d0 = (w * 12 + j) * 16;
        v8f acc0 = {}; v8f acc1 = {};
        for (int k = 0; k < H_; k += 32) {
            const v16bf af0 = ldsA(ar0, k, hi);
            const v16bf af1 = ldsA(ar1, k, hi);
            const float4* bdp = (const float4*)(wd + (size_t)(d0 + lo) * H_ + k + hi * 8);
            const v16bf bd = cvt16(bdp[0], bdp[1], bdp[4], bdp[5]);
            acc0 = WMMA_BF16(af0, bd, acc0);
            acc1 = WMMA_BF16(af1, bd, acc1);
        }
        #pragma unroll
        for (int r = 0; r < 8; ++r) {
            const int M0 = r + hi * 8;
            atomicAdd(out + (size_t)s_tok[M0]      * D_ + d0 + lo, acc0[r]);
            atomicAdd(out + (size_t)s_tok[16 + M0] * D_ + d0 + lo, acc1[r]);
        }
    }
}

// ============================== launcher ======================================
extern "C" void kernel_launch(void* const* d_in, const int* in_sizes, int n_in,
                              void* d_out, int out_size, void* d_ws, size_t ws_size,
                              hipStream_t stream) {
    const float* x        = (const float*)d_in[0];   // [B,L,D]
    const float* w_router = (const float*)d_in[1];   // [E,D]
    const float* w_gate   = (const float*)d_in[2];   // [E,H,D]
    const float* w_up     = (const float*)d_in[3];   // [E,H,D]
    const float* w_down   = (const float*)d_in[4];   // [E,D,H]
    float* out = (float*)d_out;                      // [B,L,D]

    char* ws = (char*)d_ws;
    int*   counts  = (int*)ws;                                      // E ints
    int*   tok_ids = (int*)(ws + 256);                              // E*CAP ints
    float* gw      = (float*)(ws + 256 + (size_t)E_ * CAP * 4);     // E*CAP floats

    const int nOut = NTOK * D_;
    moe_zero<<<(nOut + 255) / 256, 256, 0, stream>>>(out, counts, nOut);
    moe_router<<<NTOK / 8, 256, 0, stream>>>(x, w_router, counts, tok_ids, gw);
    dim3 grid(CAP / TT, E_);   // token-tiles x experts; expert-major for L2 reuse
    moe_expert<<<grid, 256, 0, stream>>>(x, w_gate, w_up, w_down,
                                         counts, tok_ids, gw, out);
}
